// Attention_80960133529966
// MI455X (gfx1250) — matrix operand order repaired and checked by reading, not compiled, not run on hardware
//
#include <hip/hip_runtime.h>

typedef __attribute__((ext_vector_type(16))) _Float16 v16h;
typedef __attribute__((ext_vector_type(8)))  float    v8f;

#define NUM_HEADS     32
#define HEAD_DIM      128
#define NUM_KV_HEADS  8
#define SEQ_LEN       2048
// (1/sqrt(128)) * log2(e): softmax done in base-2 so v_exp_f32 is used natively
#define SCALE_LOG2E   0.1275174513176583f
#define NEG_BIG       -1.0e30f
#define BLOCK_M       64
#define BLOCK_N       32
#define THREADS       128

// Native base-2 exponential (v_exp_f32).
__device__ __forceinline__ float fast_exp2(float x) {
    return __builtin_amdgcn_exp2f(x);
}

// Online-softmax update for one 16x32 score block held as two 16x16 C tiles.
// C layout: lane n=lane&15 holds column n; VGPR slot r = row (r + 8*hi).
template <bool MASK>
__device__ __forceinline__ void softmax_update(
    v8f& c0, v8f& c1, float* mrow, float* lpart, float* alpha,
    int qWaveBase, int kBase, int mloc, int hi)
{
#pragma unroll
    for (int r = 0; r < 8; ++r) {
        float s0 = c0[r] * SCALE_LOG2E;
        float s1 = c1[r] * SCALE_LOG2E;
        if (MASK) {
            const int rowg = qWaveBase + 8 * hi + r;
            s0 = (kBase + mloc      <= rowg) ? s0 : NEG_BIG;
            s1 = (kBase + 16 + mloc <= rowg) ? s1 : NEG_BIG;
        }
        float mx = fmaxf(s0, s1);
        mx = fmaxf(mx, __shfl_xor(mx, 1, 32));
        mx = fmaxf(mx, __shfl_xor(mx, 2, 32));
        mx = fmaxf(mx, __shfl_xor(mx, 4, 32));
        mx = fmaxf(mx, __shfl_xor(mx, 8, 32));
        const float mn = fmaxf(mrow[r], mx);
        const float p0 = fast_exp2(s0 - mn);
        const float p1 = fast_exp2(s1 - mn);
        const float a  = fast_exp2(mrow[r] - mn);
        lpart[r] = lpart[r] * a + (p0 + p1);   // per-lane partial; reduced in epilogue
        mrow[r]  = mn;
        alpha[r] = a;
        c0[r] = p0;   // reuse score regs as probabilities
        c1[r] = p1;
    }
}

// Flash-attention forward, causal, GQA (4 query heads per kv head).
// One block = 4 waves; wave w owns query rows [qt*64 + w*16, +16).
__global__ __launch_bounds__(THREADS) void fa_fwd_kernel(
    const float* __restrict__ q, const float* __restrict__ k,
    const float* __restrict__ v, float* __restrict__ out)
{
    __shared__ alignas(16) _Float16 sK [BLOCK_N * HEAD_DIM];   // [key][d]  row-major
    __shared__ alignas(16) _Float16 sVt[HEAD_DIM * BLOCK_N];   // [d][key]  transposed
    __shared__ alignas(16) _Float16 sP [4][16 * BLOCK_N];      // per-wave P scratch [m][k]

    const int tid  = threadIdx.x;
    const int wave = tid >> 5;
    const int lane = tid & 31;
    const int mloc = lane & 15;   // column (B/C) or row (A) index within 16
    const int hi   = lane >> 4;   // which half of the wave

    const int qt  = blockIdx.x;
    const int h   = blockIdx.y;
    const int b   = blockIdx.z;
    const int kvh = h >> 2;       // GQA: 4 q-heads share one kv-head

    const int qWaveBase = qt * BLOCK_M + wave * 16;

    const size_t ldq = (size_t)NUM_HEADS * HEAD_DIM;      // 4096
    const size_t ldk = (size_t)NUM_KV_HEADS * HEAD_DIM;   // 1024
    const float* qp = q   + (size_t)b * SEQ_LEN * ldq + (size_t)h   * HEAD_DIM;
    const float* kp = k   + (size_t)b * SEQ_LEN * ldk + (size_t)kvh * HEAD_DIM;
    const float* vp = v   + (size_t)b * SEQ_LEN * ldk + (size_t)kvh * HEAD_DIM;
    float*       op = out + (size_t)b * SEQ_LEN * ldq + (size_t)h   * HEAD_DIM;

    // ---- Q tile (16x128 fp32 -> f16) into 4 A-fragments (K-chunks of 32) ----
    // A layout (16-bit 16x32): lane holds row m=lane&15; half j -> K = (j&7) + 8*hi + 16*(j>>3)
    v16h qa[4];
    {
        const float* qrow = qp + (size_t)(qWaveBase + mloc) * ldq;
#pragma unroll
        for (int c = 0; c < 4; ++c) {
#pragma unroll
            for (int j = 0; j < 16; j += 2) {
                const int kk = (j & 7) + 8 * hi + ((j >> 3) << 4);
                float2 f = *(const float2*)(qrow + c * 32 + kk);
                qa[c][j]     = (_Float16)f.x;
                qa[c][j + 1] = (_Float16)f.y;
            }
        }
    }

    // ---- Output accumulators: 8 d-tiles of 16x16 f32 ----
    v8f acc[8];
#pragma unroll
    for (int dt = 0; dt < 8; ++dt)
#pragma unroll
        for (int r = 0; r < 8; ++r) acc[dt][r] = 0.0f;

    float mrow[8], lpart[8];
#pragma unroll
    for (int r = 0; r < 8; ++r) { mrow[r] = NEG_BIG; lpart[r] = 0.0f; }

    const int nSteps = (qt + 1) * (BLOCK_M / BLOCK_N);  // causal: keys up to end of q block

    for (int step = 0; step < nSteps; ++step) {
        const int kBase = step * BLOCK_N;

        __syncthreads();  // previous iteration done reading sK/sVt
        {
            // K stage: 4 threads per key row, 32 d each; packed 8B LDS stores.
            const int row   = tid >> 2;
            const int dBase = (tid & 3) * 32;
            const float* kr = kp + (size_t)(kBase + row) * ldk + dBase;
#pragma unroll
            for (int i = 0; i < 8; ++i) {
                float4 f = *(const float4*)(kr + i * 4);
                union { _Float16 hh[4]; unsigned long long u; } pk;
                pk.hh[0] = (_Float16)f.x; pk.hh[1] = (_Float16)f.y;
                pk.hh[2] = (_Float16)f.z; pk.hh[3] = (_Float16)f.w;
                *(unsigned long long*)&sK[row * HEAD_DIM + dBase + i * 4] = pk.u;
            }
        }
        {
            // V stage, d-major: thread owns column d=tid; 32 lane-coalesced b32 loads,
            // written transposed as 4 x ds_store_b128 (instead of 32 x ds_store_b16).
            const int dcol = tid;
            const float* vcol = vp + (size_t)kBase * ldk + dcol;
#pragma unroll
            for (int i = 0; i < 4; ++i) {
                union { _Float16 hh[8]; uint4 u; } pk;
#pragma unroll
                for (int j = 0; j < 8; ++j)
                    pk.hh[j] = (_Float16)vcol[(size_t)(i * 8 + j) * ldk];
                *(uint4*)&sVt[dcol * BLOCK_N + i * 8] = pk.u;
            }
        }
        if (step + 1 < nSteps) {
            // Pull next K/V tile toward L2/L0 while this step computes.
            const int kB2 = kBase + BLOCK_N;
            __builtin_prefetch(kp + (size_t)(kB2 + (tid >> 2)) * ldk + (tid & 3) * 32, 0, 1);
            __builtin_prefetch(vp + (size_t)(kB2 + (tid & 31)) * ldk + (tid >> 5) * 32, 0, 1);
        }
        __syncthreads();

        // Wave-uniform skip: this wave's rows are entirely above the diagonal.
        if (kBase > qWaveBase + 15) continue;

        // ---- S = Q * K^T : two 16x16 tiles over 4 K-chunks ----
        // B layout (16-bit 32x16): lane holds col n=lane&15; half j -> k = (j&7) + 8*hi + 16*(j>>3)
        v8f c0, c1;
#pragma unroll
        for (int r = 0; r < 8; ++r) { c0[r] = 0.0f; c1[r] = 0.0f; }
#pragma unroll
        for (int c = 0; c < 4; ++c) {
            v16h b0, b1;
#pragma unroll
            for (int j = 0; j < 16; j += 2) {
                const int kk = (j & 7) + 8 * hi + ((j >> 3) << 4);
                union { unsigned u; _Float16 hh[2]; } w0, w1;
                w0.u = *(const unsigned*)&sK[(mloc)      * HEAD_DIM + c * 32 + kk];
                w1.u = *(const unsigned*)&sK[(16 + mloc) * HEAD_DIM + c * 32 + kk];
                b0[j] = w0.hh[0]; b0[j + 1] = w0.hh[1];
                b1[j] = w1.hh[0]; b1[j + 1] = w1.hh[1];
            }
            c0 = __builtin_amdgcn_wmma_f32_16x16x32_f16(false, qa[c], false, b0, (short)0, c0, false, false);
            c1 = __builtin_amdgcn_wmma_f32_16x16x32_f16(false, qa[c], false, b1, (short)0, c1, false, false);
        }

        // ---- Online softmax (fp32, base-2). Mask only on boundary steps. ----
        float alpha[8];
        if (kBase + 31 > qWaveBase)
            softmax_update<true >(c0, c1, mrow, lpart, alpha, qWaveBase, kBase, mloc, hi);
        else
            softmax_update<false>(c0, c1, mrow, lpart, alpha, qWaveBase, kBase, mloc, hi);

        // Rescale accumulated output rows
#pragma unroll
        for (int dt = 0; dt < 8; ++dt)
#pragma unroll
            for (int r = 0; r < 8; ++r) acc[dt][r] *= alpha[r];

        // ---- Re-layout P (C layout -> A layout) via per-wave LDS scratch ----
        _Float16* pw = &sP[wave][0];
#pragma unroll
        for (int r = 0; r < 8; ++r) {
            const int mm = 8 * hi + r;
            pw[mm * BLOCK_N + mloc]      = (_Float16)c0[r];
            pw[mm * BLOCK_N + 16 + mloc] = (_Float16)c1[r];
        }
        asm volatile("s_wait_dscnt 0" ::: "memory");   // wave-local store->load ordering

        v16h pa;
#pragma unroll
        for (int j = 0; j < 16; j += 2) {
            const int kk = (j & 7) + 8 * hi + ((j >> 3) << 4);  // A layout
            union { unsigned u; _Float16 hh[2]; } w;
            w.u = *(const unsigned*)&pw[mloc * BLOCK_N + kk];
            pa[j] = w.hh[0]; pa[j + 1] = w.hh[1];
        }

        // ---- O += P * V : 8 d-tiles; B from transposed V, k = (j&7) + 8*hi + 16*(j>>3) ----
#pragma unroll
        for (int dt = 0; dt < 8; ++dt) {
            v16h bv;
#pragma unroll
            for (int j = 0; j < 16; j += 2) {
                const int kk = (j & 7) + 8 * hi + ((j >> 3) << 4);
                union { unsigned u; _Float16 hh[2]; } w;
                w.u = *(const unsigned*)&sVt[(dt * 16 + mloc) * BLOCK_N + kk];
                bv[j] = w.hh[0]; bv[j + 1] = w.hh[1];
            }
            acc[dt] = __builtin_amdgcn_wmma_f32_16x16x32_f16(false, pa, false, bv, (short)0, acc[dt], false, false);
        }
    }

    // ---- Epilogue: reduce deferred row sums, normalize, store O (fp32) ----
#pragma unroll
    for (int r = 0; r < 8; ++r) {
        float l = lpart[r];
        l += __shfl_xor(l, 1, 32);
        l += __shfl_xor(l, 2, 32);
        l += __shfl_xor(l, 4, 32);
        l += __shfl_xor(l, 8, 32);
        const float inv = 1.0f / l;
        const int rowg = qWaveBase + 8 * hi + r;
        float* orow = op + (size_t)rowg * ldq;
#pragma unroll
        for (int dt = 0; dt < 8; ++dt)
            orow[dt * 16 + mloc] = acc[dt][r] * inv;
    }
}

extern "C" void kernel_launch(void* const* d_in, const int* in_sizes, int n_in,
                              void* d_out, int out_size, void* d_ws, size_t ws_size,
                              hipStream_t stream) {
    const float* q = (const float*)d_in[0];
    const float* k = (const float*)d_in[1];
    const float* v = (const float*)d_in[2];
    float* out = (float*)d_out;
    (void)n_in; (void)out_size; (void)d_ws; (void)ws_size;

    const int T = in_sizes[0] / (NUM_HEADS * HEAD_DIM);   // total tokens
    const int B = T / SEQ_LEN;                            // batch (reference: seq_len = 2048)
    dim3 grid(SEQ_LEN / BLOCK_M, NUM_HEADS, B);
    fa_fwd_kernel<<<grid, THREADS, 0, stream>>>(q, k, v, out);
}
